// MultiHeadedAttention_33964601377494
// MI455X (gfx1250) — matrix operand order repaired and checked by reading, not compiled, not run on hardware
//
#include <hip/hip_runtime.h>

#define B_  2
#define S_  2048
#define D_  768
#define H_  12
#define DK_ 64

typedef __attribute__((ext_vector_type(16))) __bf16 v16bf;
typedef __attribute__((ext_vector_type(8)))  float  v8f;
typedef __attribute__((ext_vector_type(4)))  float  v4f;

__device__ __forceinline__ v8f wmma_bf16(v16bf a, v16bf b, v8f c) {
  // (neg_a, A, neg_b, B, c_mod, C, reuse_a, reuse_b)
  return __builtin_amdgcn_wmma_f32_16x16x32_bf16(false, a, false, b, (short)0, c,
                                                 false, false);
}

// A fragment 16x32 (MxK): lane holds row m = lane&15, hi = lane>>4.
// element e -> k = e + 8*hi + (e & 8). 'row' points at src[m][k0].
__device__ __forceinline__ v16bf frag_a_f32(const float* row, int hi) {
  v16bf f;
#pragma unroll
  for (int e = 0; e < 8; ++e) {
    f[e]     = (__bf16)row[8 * hi + e];
    f[e + 8] = (__bf16)row[16 + 8 * hi + e];
  }
  return f;
}
__device__ __forceinline__ v16bf frag_a_bf16(const __bf16* row, int hi) {
  v16bf f;
#pragma unroll
  for (int e = 0; e < 8; ++e) {
    f[e]     = row[8 * hi + e];
    f[e + 8] = row[16 + 8 * hi + e];
  }
  return f;
}
// B fragment 32x16 (KxN): lane holds col n = lane&15; element e -> k = e + 8*hi + (e & 8).
// 'row' points at src[n][k0] for a source stored [n][k] (i.e. B pre-transposed).
__device__ __forceinline__ v16bf frag_b_f32(const float* row, int hi) {
  v16bf f;
#pragma unroll
  for (int e = 0; e < 16; ++e) f[e] = (__bf16)row[8 * hi + e + (e & 8)];
  return f;
}
__device__ __forceinline__ v16bf frag_b_bf16(const __bf16* row, int hi) {
  v16bf f;
#pragma unroll
  for (int e = 0; e < 16; ++e) f[e] = row[8 * hi + e + (e & 8)];
  return f;
}

// ---------------- Kernel 1: Q/K/V projections ----------------
// Y = X @ W^T + b  (N=4096 rows, K=768, J=768), bf16 WMMA, fp32 accum.
// Q,K stored [B,H,S,DK]; V stored transposed [B,H,DK,S].
__global__ __launch_bounds__(128) void mha_qkv_proj(
    const float* __restrict__ X0, const float* __restrict__ X1,
    const float* __restrict__ X2,
    const float* __restrict__ Wq, const float* __restrict__ bq,
    const float* __restrict__ Wk, const float* __restrict__ bk,
    const float* __restrict__ Wv, const float* __restrict__ bv,
    __bf16* __restrict__ qbf, __bf16* __restrict__ kbf,
    __bf16* __restrict__ vtbf) {
  const int wave = threadIdx.x >> 5, lane = threadIdx.x & 31;
  const int ln = lane & 15, hi = lane >> 4;
  const int tile = blockIdx.x * 4 + wave;          // 3 * 256 * 48 = 36864 tiles
  const int p   = tile / (256 * 48);
  const int rem = tile % (256 * 48);
  const int nt = rem / 48, jt = rem % 48;

  const float* X = (p == 0) ? X0 : (p == 1) ? X1 : X2;
  const float* W = (p == 0) ? Wq : (p == 1) ? Wk : Wv;
  const float* bias = (p == 0) ? bq : (p == 1) ? bk : bv;

  const int m_g = nt * 16 + ln;  // A row for this lane
  const int j_g = jt * 16 + ln;  // C/D column (and W row) for this lane

  v8f acc = {};
  for (int kt = 0; kt < D_ / 32; ++kt) {
    v16bf a  = frag_a_f32(X + (size_t)m_g * D_ + kt * 32, hi);
    v16bf bb = frag_b_f32(W + (size_t)j_g * D_ + kt * 32, hi);  // B[k][j] = W[j][k]
    acc = wmma_bf16(a, bb, acc);
  }
  const float bj = bias[j_g];
  const int h = j_g >> 6, dk = j_g & 63;
#pragma unroll
  for (int i = 0; i < 8; ++i) {
    const int n_g = nt * 16 + i + 8 * hi;
    const int s = n_g & (S_ - 1), bidx = n_g >> 11;
    const int bh = bidx * H_ + h;
    const __bf16 v = (__bf16)(acc[i] + bj);
    if (p == 0)      qbf [((size_t)bh * S_ + s) * DK_ + dk] = v;
    else if (p == 1) kbf [((size_t)bh * S_ + s) * DK_ + dk] = v;
    else             vtbf[((size_t)bh * DK_ + dk) * S_ + s] = v;
  }
}

// ---------------- Kernel 2: fused attention ----------------
// One block = one (b,h,16-query-row) strip. Score strip [16 x 2048] f32 in LDS,
// single-exp softmax (P' = exp(s - rowmax), O rows rescaled by 1/Z at the end),
// log_attn streamed out with non-temporal b128 stores, O = P' @ V via WMMA.
#define SMEM_STRIP  (16 * S_ * 4)       // 131072 B
#define SMEM_PBF    (16 * S_ * 2)       // 65536 B
#define SMEM_BYTES  (SMEM_STRIP + SMEM_PBF + 128 * 4 + 3 * 16 * 4)

__global__ __launch_bounds__(128) void mha_attn(
    const __bf16* __restrict__ qbf, const __bf16* __restrict__ kbf,
    const __bf16* __restrict__ vtbf, const int* __restrict__ mask,
    float* __restrict__ log_attn, float* __restrict__ ao) {
  extern __shared__ char smem[];
  float*  strip  = (float*)smem;                         // [16][2048] f32
  __bf16* pbf    = (__bf16*)(smem + SMEM_STRIP);         // [16][2048] bf16
  float*  red    = (float*)(smem + SMEM_STRIP + SMEM_PBF);
  float*  rowmax = red + 128;
  float*  rowlz  = rowmax + 16;
  float*  rowrcp = rowlz + 16;

  const int t = threadIdx.x;
  const int wave = t >> 5, lane = t & 31;
  const int ln = lane & 15, hi = lane >> 4;

  const int qt = blockIdx.x & 127;
  const int bh = blockIdx.x >> 7;
  const int b  = bh / H_;
  const int h  = bh % H_;
  const int q0 = qt * 16;

  const __bf16* qbase = qbf  + (size_t)bh * S_ * DK_;
  const __bf16* kbase = kbf  + (size_t)bh * S_ * DK_;
  const __bf16* vbase = vtbf + (size_t)bh * DK_ * S_;

  // ---- scores strip: scale * Q·K^T, masked ----
  for (int kt2 = wave; kt2 < S_ / 16; kt2 += 4) {
    if (kt2 + 4 < S_ / 16)  // prefetch next K tile (global_prefetch_b8)
      __builtin_prefetch(kbase + (size_t)((kt2 + 4) * 16 + ln) * DK_, 0, 3);
    v8f acc = {};
#pragma unroll
    for (int dh = 0; dh < 2; ++dh) {  // DK = 64 -> two K=32 steps
      v16bf a  = frag_a_bf16(qbase + (size_t)(q0 + ln) * DK_ + dh * 32, hi);
      v16bf bb = frag_b_bf16(kbase + (size_t)(kt2 * 16 + ln) * DK_ + dh * 32, hi);
      acc = wmma_bf16(a, bb, acc);
    }
    const int kk = kt2 * 16 + ln;
#pragma unroll
    for (int i = 0; i < 8; ++i) {
      const int r = i + 8 * hi;
      float sc = acc[i] * 0.125f;  // 1/sqrt(64)
      const int mv =
          __builtin_nontemporal_load(&mask[((size_t)b * S_ + (q0 + r)) * S_ + kk]);
      if (mv == 0) sc = -1e9f;
      strip[r * S_ + kk] = sc;
    }
  }
  __syncthreads();

  // ---- pass 1: row max (8 threads per row, 256 elements each) ----
  const int r = t >> 3, c = t & 7;
  const int base = r * S_ + c * 256;
  float mx = -3.4e38f;
  for (int i = 0; i < 256; ++i) mx = fmaxf(mx, strip[base + i]);
  red[t] = mx;
  __syncthreads();
  if (t < 16) {
    float m2 = red[t * 8];
#pragma unroll
    for (int j = 1; j < 8; ++j) m2 = fmaxf(m2, red[t * 8 + j]);
    rowmax[t] = m2;
  }
  __syncthreads();

  // ---- pass 2 (single exp): P' = exp(s - rowmax) -> bf16 LDS, accumulate Z ----
  const float rm = rowmax[r];
  float sm = 0.f;
  for (int i = 0; i < 256; i += 2) {
    const float e0 = __expf(strip[base + i]     - rm);
    const float e1 = __expf(strip[base + i + 1] - rm);
    sm += e0 + e1;
    pbf[base + i]     = (__bf16)e0;
    pbf[base + i + 1] = (__bf16)e1;
  }
  red[t] = sm;
  __syncthreads();
  if (t < 16) {
    float Z = 0.f;
#pragma unroll
    for (int j = 0; j < 8; ++j) Z += red[t * 8 + j];
    rowlz[t]  = rowmax[t] + __logf(Z);  // logsumexp (exact for log_attn)
    rowrcp[t] = 1.0f / Z;               // folded into O rows after WMMA
  }
  __syncthreads();

  // ---- emit log_attn: 402 MB stream, vectorized non-temporal b128 stores ----
  {
    v4f* la4 = (v4f*)(log_attn + ((size_t)bh * S_ + q0) * S_);
    const v4f* st4 = (const v4f*)strip;
    for (int i4 = t; i4 < (16 * S_) / 4; i4 += 128) {
      const float lz = rowlz[i4 >> 9];        // 512 v4f per row
      v4f o = st4[i4] - lz;
      __builtin_nontemporal_store(o, la4 + i4);
    }
  }
  __syncthreads();

  // ---- O[16x64] = (P'[16x2048] @ V[2048x64]) * (1/Z per row) ----
  {
    v8f acc = {};
    const __bf16* vrow = vbase + (size_t)(wave * 16 + ln) * S_;  // V^T row = col dk
    for (int kt = 0; kt < S_ / 32; ++kt) {
      if (kt + 8 < S_ / 32)  // prefetch next V^T chunk
        __builtin_prefetch(vrow + (kt + 8) * 32, 0, 3);
      v16bf a  = frag_a_bf16(pbf + ln * S_ + kt * 32, hi);
      v16bf bb = frag_b_bf16(vrow + kt * 32, hi);
      acc = wmma_bf16(a, bb, acc);
    }
#pragma unroll
    for (int i = 0; i < 8; ++i) {
      const int rr = i + 8 * hi;
      ao[((size_t)b * S_ + q0 + rr) * D_ + h * DK_ + wave * 16 + ln] =
          acc[i] * rowrcp[rr];
    }
  }
}

// ---------------- Kernel 3: output projection ----------------
__global__ __launch_bounds__(128) void mha_out_proj(
    const float* __restrict__ ao, const float* __restrict__ Wo,
    const float* __restrict__ bo, float* __restrict__ out) {
  const int wave = threadIdx.x >> 5, lane = threadIdx.x & 31;
  const int ln = lane & 15, hi = lane >> 4;
  const int tile = blockIdx.x * 4 + wave;  // 256 * 48 tiles
  const int nt = tile / 48, jt = tile % 48;
  const int m_g = nt * 16 + ln;
  const int j_g = jt * 16 + ln;

  v8f acc = {};
  for (int kt = 0; kt < D_ / 32; ++kt) {
    v16bf a  = frag_a_f32(ao + (size_t)m_g * D_ + kt * 32, hi);
    v16bf bb = frag_b_f32(Wo + (size_t)j_g * D_ + kt * 32, hi);
    acc = wmma_bf16(a, bb, acc);
  }
  const float bj = bo[j_g];
#pragma unroll
  for (int i = 0; i < 8; ++i) {
    const int n_g = nt * 16 + i + 8 * hi;
    out[(size_t)n_g * D_ + j_g] = acc[i] + bj;
  }
}

extern "C" void kernel_launch(void* const* d_in, const int* in_sizes, int n_in,
                              void* d_out, int out_size, void* d_ws, size_t ws_size,
                              hipStream_t stream) {
  const float* query = (const float*)d_in[0];
  const float* key_t = (const float*)d_in[1];
  const float* value = (const float*)d_in[2];
  const int*   mask  = (const int*)d_in[3];
  const float* Wq = (const float*)d_in[4];
  const float* bq = (const float*)d_in[5];
  const float* Wk = (const float*)d_in[6];
  const float* bk = (const float*)d_in[7];
  const float* Wv = (const float*)d_in[8];
  const float* bv = (const float*)d_in[9];
  const float* Wo = (const float*)d_in[10];
  const float* bo = (const float*)d_in[11];
  float* out = (float*)d_out;

  // workspace layout (bytes): qbf 6MB | kbf 6MB | vtbf 6MB | ao 12MB  (~30 MB)
  char* ws = (char*)d_ws;
  __bf16* qbf  = (__bf16*)(ws);
  __bf16* kbf  = (__bf16*)(ws + 6291456);
  __bf16* vtbf = (__bf16*)(ws + 12582912);
  float*  ao   = (float*)(ws + 18874368);

  (void)in_sizes; (void)n_in; (void)out_size; (void)ws_size;

  // allow >64KB dynamic LDS (197 KB strip; WGP supports 320 KB)
  hipFuncSetAttribute((const void*)mha_attn,
                      hipFuncAttributeMaxDynamicSharedMemorySize, SMEM_BYTES);

  // 1) QKV projections: 36864 tiles / 4 waves per block
  mha_qkv_proj<<<9216, 128, 0, stream>>>(query, key_t, value,
                                         Wq, bq, Wk, bk, Wv, bv,
                                         qbf, kbf, vtbf);
  // 2) fused attention: B*H*(S/16) = 3072 blocks; log_attn after `out`
  mha_attn<<<3072, 128, SMEM_BYTES, stream>>>(qbf, kbf, vtbf, mask,
                                              out + (size_t)B_ * S_ * D_, ao);
  // 3) output projection: 12288 tiles / 4 waves per block
  mha_out_proj<<<3072, 128, 0, stream>>>(ao, Wo, bo, out);
}
